// ShiftedWindowAttention_22144851378718
// MI455X (gfx1250) — compile-verified, matrix operand order checked by reading, not run on hardware
//
#include <hip/hip_runtime.h>

// ---------------------------------------------------------------------------
// Fused Swin shifted-window attention for gfx1250 (MI455X).
// One workgroup (256 threads = 8 wave32) per 7x7 window. All intermediates
// (f16 q/k/v, softmax probs, head outputs) live in the 320KB WGP LDS; the
// only HBM traffic is x in / out, weights stream from L2 in f16.
// All GEMMs use v_wmma_f32_16x16x32_f16 with T padded 49 -> 64.
// ---------------------------------------------------------------------------

typedef __attribute__((ext_vector_type(16))) _Float16 v16h;
typedef __attribute__((ext_vector_type(8)))  _Float16 v8h;
typedef __attribute__((ext_vector_type(8)))  float    v8f;

#define LDQ 400   // row stride (halves) for 64x384 activation tiles (+16 pad)
#define LDT 72    // row stride (halves) for transposed-v [384][64] (+8 pad)
#define LDP 72    // row stride (halves) for per-wave 64x64 prob tile

// LDS layout (offsets in halves)
#define XS_OFF 0                         // x window / head-output O: 64*LDQ
#define QS_OFF (XS_OFF + 64 * LDQ)       // q (scaled):               64*LDQ
#define KS_OFF (QS_OFF + 64 * LDQ)       // k:                        64*LDQ
#define VT_OFF (KS_OFF + 64 * LDQ)       // v transposed [c][t]:     384*LDT
#define P_OFF  (VT_OFF + 384 * LDT)      // probs, per wave:       8*64*LDP
#define SMEM_HALVES (P_OFF + 8 * 64 * LDP)
#define SMEM_BYTES  (SMEM_HALVES * 2)    // = 282624 B  (< 320KB WGP LDS)

// Load one 16x32 f16 WMMA A/B fragment for this lane.
// p = row/col base + k0 + g*8 ; halves [0..7] at p, [8..15] at p+16.
__device__ __forceinline__ v16h frag_ld(const _Float16* p) {
    v8h lo = *(const v8h*)(p);
    v8h hi = *(const v8h*)(p + 16);
    v16h r;
#pragma unroll
    for (int i = 0; i < 8; ++i) { r[i] = lo[i]; r[8 + i] = hi[i]; }
    return r;
}

__device__ __forceinline__ v8f wmma_f16(v16h a, v16h b, v8f c) {
    return __builtin_amdgcn_wmma_f32_16x16x32_f16(false, a, false, b,
                                                  (short)0, c, false, false);
}

// region-id tier for the shifted-window mask (ws=7, shift=3, pad=56)
__device__ __forceinline__ int mask_tier(int p) {
    return (p < 49) ? 0 : ((p < 53) ? 1 : 2);
}

// --------------------------- weight f32 -> f16 -----------------------------
__global__ void convert_weights(const float* __restrict__ qkvw,
                                const float* __restrict__ projw,
                                _Float16* __restrict__ wq,
                                _Float16* __restrict__ wp) {
    int i = blockIdx.x * blockDim.x + threadIdx.x;
    if (i < 1152 * 384) wq[i] = (_Float16)qkvw[i];
    if (i < 384 * 384)  wp[i] = (_Float16)projw[i];
}

// ------------------------------ fused kernel -------------------------------
__global__ void __launch_bounds__(256)
swin_fused(const float* __restrict__ x,          // [32,56,56,384]
           const float* __restrict__ qkvb,       // [1152]
           const float* __restrict__ projb,      // [384]
           const float* __restrict__ rpbt,       // [169,12]
           const _Float16* __restrict__ wqkv,    // [1152,384] f16
           const _Float16* __restrict__ wproj,   // [384,384]  f16
           float* __restrict__ out) {            // [32,56,56,384]
    extern __shared__ _Float16 sm[];

    const int blk  = blockIdx.x;
    const int b    = blk >> 6;
    const int wy   = (blk >> 3) & 7;
    const int wx   = blk & 7;
    const int tid  = threadIdx.x;
    const int wave = tid >> 5;
    const int lane = tid & 31;
    const int g    = lane >> 4;     // half-wave id
    const int l16  = lane & 15;

    // ---------------- Phase 0: load window (roll folded in), f32->f16 ------
    for (int i = tid; i < 49 * 96; i += 256) {
        int t  = i / 96;
        int c4 = (i % 96) * 4;
        int iy = t / 7, ix = t % 7;
        int sh = (wy * 7 + iy + 3) % 56;        // roll(-3) source row
        int sw = (wx * 7 + ix + 3) % 56;
        const float4 v = *(const float4*)(x + (((size_t)b * 56 + sh) * 56 + sw) * 384 + c4);
        _Float16* d = &sm[XS_OFF + t * LDQ + c4];
        d[0] = (_Float16)v.x; d[1] = (_Float16)v.y;
        d[2] = (_Float16)v.z; d[3] = (_Float16)v.w;
    }
    for (int i = tid; i < 15 * 96; i += 256) {  // zero pad rows 49..63
        int t  = 49 + i / 96;
        int c4 = (i % 96) * 4;
        _Float16* d = &sm[XS_OFF + t * LDQ + c4];
        d[0] = d[1] = d[2] = d[3] = (_Float16)0.0f;
    }
    __syncthreads();

    // ---------------- Phase 1: QKV GEMM  [64x384]x[384x1152] ---------------
    const float qscale = 0.17677669529663687f;  // dh^-0.5
    for (int j = 0; j < 9; ++j) {
        const int nt  = wave + 8 * j;           // N-tile, 72 total
        const int n0  = nt * 16;
        const int col = n0 + l16;               // output channel 0..1151
        v8f acc[4] = {};
        for (int kk = 0; kk < 12; ++kk) {
            const int k0 = kk * 32;
            v16h bf = frag_ld(wqkv + (size_t)col * 384 + k0 + g * 8);
#pragma unroll
            for (int mt = 0; mt < 4; ++mt) {
                v16h af = frag_ld(&sm[XS_OFF + (mt * 16 + l16) * LDQ + k0 + g * 8]);
                acc[mt] = wmma_f16(af, bf, acc[mt]);
            }
        }
        const float bias = qkvb[col];
        const int sec = n0 / 384;               // 0=q 1=k 2=v (tile-uniform)
        const int c   = col - sec * 384;        // h*32 + d
        if (sec == 0) {
#pragma unroll
            for (int mt = 0; mt < 4; ++mt)
#pragma unroll
                for (int r = 0; r < 8; ++r) {
                    int m = mt * 16 + g * 8 + r;
                    sm[QS_OFF + m * LDQ + c] = (_Float16)((acc[mt][r] + bias) * qscale);
                }
        } else if (sec == 1) {
#pragma unroll
            for (int mt = 0; mt < 4; ++mt)
#pragma unroll
                for (int r = 0; r < 8; ++r) {
                    int m = mt * 16 + g * 8 + r;
                    sm[KS_OFF + m * LDQ + c] = (_Float16)(acc[mt][r] + bias);
                }
        } else {                                 // v: transposed, contiguous m
#pragma unroll
            for (int mt = 0; mt < 4; ++mt) {
                v8h pk;
#pragma unroll
                for (int r = 0; r < 8; ++r) pk[r] = (_Float16)(acc[mt][r] + bias);
                *(v8h*)&sm[VT_OFF + c * LDT + mt * 16 + g * 8] = pk;
            }
        }
    }
    __syncthreads();

    // ---------------- Phase 2: attention, waves own heads ------------------
    _Float16* Pw = &sm[P_OFF + wave * (64 * LDP)];
    for (int h = wave; h < 12; h += 8) {
        const int d0 = h * 32;

        // S = q_h @ k_h^T : M=64, N=64, K=32 (single WMMA k-step)
        v16h aq[4];
#pragma unroll
        for (int mt = 0; mt < 4; ++mt)
            aq[mt] = frag_ld(&sm[QS_OFF + (mt * 16 + l16) * LDQ + d0 + g * 8]);
        v8f sacc[4][4];
#pragma unroll
        for (int ntt = 0; ntt < 4; ++ntt) {
            v16h bk = frag_ld(&sm[KS_OFF + (ntt * 16 + l16) * LDQ + d0 + g * 8]);
            v8f z = {};
#pragma unroll
            for (int mt = 0; mt < 4; ++mt)
                sacc[mt][ntt] = wmma_f16(aq[mt], bk, z);
        }

        // add relative-position bias + shifted-window mask, mask pad cols
#pragma unroll
        for (int mt = 0; mt < 4; ++mt) {
#pragma unroll
            for (int r = 0; r < 8; ++r) {
                const int m = mt * 16 + g * 8 + r;
                if (m < 49) {
                    const int ty = m / 7, tx = m % 7;
                    const int rid_t = mask_tier(wy * 7 + ty) * 3 + mask_tier(wx * 7 + tx);
#pragma unroll
                    for (int ntt = 0; ntt < 4; ++ntt) {
                        const int s = ntt * 16 + l16;
                        if (s < 49) {
                            const int sy = s / 7, sx = s % 7;
                            const float rb =
                                rpbt[((ty - sy + 6) * 13 + (tx - sx + 6)) * 12 + h];
                            const int rid_s =
                                mask_tier(wy * 7 + sy) * 3 + mask_tier(wx * 7 + sx);
                            sacc[mt][ntt][r] += rb + ((rid_t != rid_s) ? -100.0f : 0.0f);
                        } else {
                            sacc[mt][ntt][r] = -1e30f;
                        }
                    }
                } else {
#pragma unroll
                    for (int ntt = 0; ntt < 4; ++ntt)
                        if (ntt * 16 + l16 >= 49) sacc[mt][ntt][r] = -1e30f;
                }
            }
        }

        // softmax per row: row m lives in one 16-lane group, 4 n-tiles
#pragma unroll
        for (int mt = 0; mt < 4; ++mt) {
#pragma unroll
            for (int r = 0; r < 8; ++r) {
                float rm = -3e38f;
#pragma unroll
                for (int ntt = 0; ntt < 4; ++ntt) rm = fmaxf(rm, sacc[mt][ntt][r]);
#pragma unroll
                for (int off = 8; off >= 1; off >>= 1)
                    rm = fmaxf(rm, __shfl_xor(rm, off));
                float rs = 0.0f;
#pragma unroll
                for (int ntt = 0; ntt < 4; ++ntt) {
                    float e = __expf(sacc[mt][ntt][r] - rm);
                    sacc[mt][ntt][r] = e;
                    rs += e;
                }
#pragma unroll
                for (int off = 8; off >= 1; off >>= 1)
                    rs += __shfl_xor(rs, off);
                const float inv = 1.0f / rs;
                const int m = mt * 16 + g * 8 + r;
#pragma unroll
                for (int ntt = 0; ntt < 4; ++ntt)
                    Pw[m * LDP + ntt * 16 + l16] = (_Float16)(sacc[mt][ntt][r] * inv);
            }
        }

        // O = P @ v_h : M=64, N=32, K=64 (2 WMMA k-steps)
        v8f oacc[4][2] = {};
#pragma unroll
        for (int kk = 0; kk < 2; ++kk) {
            const int k0 = kk * 32;
            v16h bv[2];
#pragma unroll
            for (int dt = 0; dt < 2; ++dt)
                bv[dt] = frag_ld(&sm[VT_OFF + (d0 + dt * 16 + l16) * LDT + k0 + g * 8]);
#pragma unroll
            for (int mt = 0; mt < 4; ++mt) {
                v16h ap = frag_ld(Pw + (mt * 16 + l16) * LDP + k0 + g * 8);
#pragma unroll
                for (int dt = 0; dt < 2; ++dt)
                    oacc[mt][dt] = wmma_f16(ap, bv[dt], oacc[mt][dt]);
            }
        }
        // store head output into the (now free) x region, layout [t][h*32+d]
#pragma unroll
        for (int mt = 0; mt < 4; ++mt)
#pragma unroll
            for (int dt = 0; dt < 2; ++dt) {
                const int c = d0 + dt * 16 + l16;
#pragma unroll
                for (int r = 0; r < 8; ++r) {
                    const int m = mt * 16 + g * 8 + r;
                    sm[XS_OFF + m * LDQ + c] = (_Float16)oacc[mt][dt][r];
                }
            }
    }
    __syncthreads();

    // ---------------- Phase 3: output proj [64x384]x[384x384], store -------
    for (int j = 0; j < 3; ++j) {
        const int nt  = wave * 3 + j;           // 24 N-tiles total
        const int n0  = nt * 16;
        const int col = n0 + l16;               // 0..383
        v8f acc[4] = {};
        for (int kk = 0; kk < 12; ++kk) {
            const int k0 = kk * 32;
            v16h bf = frag_ld(wproj + (size_t)col * 384 + k0 + g * 8);
#pragma unroll
            for (int mt = 0; mt < 4; ++mt) {
                v16h af = frag_ld(&sm[XS_OFF + (mt * 16 + l16) * LDQ + k0 + g * 8]);
                acc[mt] = wmma_f16(af, bf, acc[mt]);
            }
        }
        const float pb = projb[col];
#pragma unroll
        for (int mt = 0; mt < 4; ++mt)
#pragma unroll
            for (int r = 0; r < 8; ++r) {
                const int m = mt * 16 + g * 8 + r;
                if (m < 49) {
                    const int iy = m / 7, ix = m % 7;
                    const int sh = (wy * 7 + iy + 3) % 56;   // roll(+3) target
                    const int sw = (wx * 7 + ix + 3) % 56;
                    out[(((size_t)b * 56 + sh) * 56 + sw) * 384 + col] = acc[mt][r] + pb;
                }
            }
    }
}

// ------------------------------- launcher ----------------------------------
extern "C" void kernel_launch(void* const* d_in, const int* in_sizes, int n_in,
                              void* d_out, int out_size, void* d_ws, size_t ws_size,
                              hipStream_t stream) {
    const float* x     = (const float*)d_in[0];
    const float* qkvw  = (const float*)d_in[1];
    const float* qkvb  = (const float*)d_in[2];
    const float* projw = (const float*)d_in[3];
    const float* projb = (const float*)d_in[4];
    const float* rpbt  = (const float*)d_in[5];
    float* out = (float*)d_out;

    _Float16* wq = (_Float16*)d_ws;            // 1152*384 f16
    _Float16* wp = wq + 1152 * 384;            //  384*384 f16

    convert_weights<<<(1152 * 384 + 255) / 256, 256, 0, stream>>>(qkvw, projw, wq, wp);

    (void)hipFuncSetAttribute((const void*)swin_fused,
                              hipFuncAttributeMaxDynamicSharedMemorySize,
                              SMEM_BYTES);
    // 32 batches * 64 windows = 2048 workgroups, one per window
    swin_fused<<<2048, 256, SMEM_BYTES, stream>>>(x, qkvb, projb, rpbt, wq, wp, out);
}
